// Pointer_network_19473381720391
// MI455X (gfx1250) — compile-verified, matrix operand order checked by reading, not run on hardware
//
#include <hip/hip_runtime.h>
#include <hip/hip_bf16.h>
#include <math.h>
#include <stdint.h>

// Problem constants (from reference)
#define BB   512    // batch
#define SS   50     // sequence length
#define EE   128    // embedding dim
#define HH   128    // hidden dim
#define G4   512    // 4*H gates
#define NEGV (-1e9f)
#define CEXP 10.0f

typedef __attribute__((ext_vector_type(16))) _Float16 v16h;
typedef __attribute__((ext_vector_type(8)))  float    v8f;

#define DEV __device__ __forceinline__

DEV float sigm(float x) { return 1.0f / (1.0f + __expf(-x)); }

// deterministic counter hash -> (0,1)
DEV float u01(uint32_t x) {
  x ^= x >> 17; x *= 0xed5ad4bbu;
  x ^= x >> 11; x *= 0xac4c1b51u;
  x ^= x >> 15; x *= 0x31848babu;
  x ^= x >> 14;
  return ((x >> 8) + 0.5f) * (1.0f / 16777216.0f);
}

// CDNA5 16-bit A/B fragment loader for V_WMMA_*_16X16X32.
// Per ISA 7.12.2: lanes 0-15 hold K={0..7,16..23}, lanes 16-31 hold K={8..15,24..31},
// packed as consecutive pairs per VGPR. rowbase points at (row, k0) of a row-major
// matrix; g = lane>>4.
DEV v16h frag_row(const _Float16* rowbase, int g) {
  v16h v;
#pragma unroll
  for (int j = 0; j < 16; ++j)
    v[j] = rowbase[(j & 7) + ((j >> 3) << 4) + (g << 3)];
  return v;
}

// ---------------------------------------------------------------------------
// prep kernels
// ---------------------------------------------------------------------------
__global__ void f2h_kernel(const float* __restrict__ src, _Float16* __restrict__ dst, int n) {
  int i = blockIdx.x * blockDim.x + threadIdx.x;
  if (i < n) dst[i] = (_Float16)src[i];
}

// emb[b][s][e] = input[b][0][s]*E[0][e] + input[b][1][s]*E[1][e]
__global__ void emb_kernel(const float* __restrict__ input, const float* __restrict__ Ew,
                           _Float16* __restrict__ emb) {
  int i = blockIdx.x * blockDim.x + threadIdx.x;
  if (i >= BB * SS * EE) return;
  int e = i & (EE - 1), bs = i >> 7;
  int s = bs % SS, b = bs / SS;
  float v = input[b * 2 * SS + s] * Ew[e] + input[b * 2 * SS + SS + s] * Ew[EE + e];
  emb[i] = (_Float16)v;
}

__global__ void init_kernel(const float* __restrict__ g0, _Float16* __restrict__ h_h,
                            float* __restrict__ c, _Float16* __restrict__ x_h,
                            unsigned char* __restrict__ mask, int* __restrict__ idx) {
  int i = blockIdx.x * blockDim.x + threadIdx.x;
  if (i < BB * HH) {
    h_h[i] = (_Float16)0.f;
    c[i] = 0.f;
    x_h[i] = (_Float16)g0[i & (EE - 1)];
  }
  if (i < BB * SS) mask[i] = 0;
  if (i < BB) idx[i] = 0;
}

// ---------------------------------------------------------------------------
// fused LSTM step: gates = x@Wih^T + h@Whh^T (WMMA, f16 in / f32 acc), then
// pointwise LSTM. One block = 16 batch rows, 4 waves cover all 32 gate N-tiles.
// ---------------------------------------------------------------------------
__global__ __launch_bounds__(128)
void lstm_step_kernel(const _Float16* __restrict__ x, int x_stride,
                      const _Float16* __restrict__ Wih,   // 512x128 f16 row-major [out][k]
                      const _Float16* __restrict__ Whh,   // 512x128
                      const float* __restrict__ bih, const float* __restrict__ bhh,
                      float* __restrict__ c, _Float16* __restrict__ h_h,
                      _Float16* __restrict__ enc_out, int enc_stride) {
  __shared__ _Float16 sx[16 * EE];
  __shared__ _Float16 sh[16 * HH];
  __shared__ float    sg[16 * G4];
  const int tid = threadIdx.x;
  const int lane = tid & 31, wave = tid >> 5;
  const int b0 = blockIdx.x * 16;
  for (int i = tid; i < 16 * EE; i += 128) {
    int r = i >> 7, cc = i & 127;
    sx[i] = x[(b0 + r) * x_stride + cc];
    sh[i] = h_h[(b0 + r) * HH + cc];
  }
  __syncthreads();
  const int g = lane >> 4, m = lane & 15;
#pragma unroll 1
  for (int nt = wave * 8; nt < wave * 8 + 8; ++nt) {
    v8f acc = {};
#pragma unroll
    for (int kt = 0; kt < 4; ++kt) {
      v16h a = frag_row(sx + m * EE + kt * 32, g);
      v16h b = frag_row(Wih + (nt * 16 + m) * EE + kt * 32, g);
      acc = __builtin_amdgcn_wmma_f32_16x16x32_f16(false, a, false, b, (short)0, acc, false, false);
    }
#pragma unroll
    for (int kt = 0; kt < 4; ++kt) {
      v16h a = frag_row(sh + m * HH + kt * 32, g);
      v16h b = frag_row(Whh + (nt * 16 + m) * HH + kt * 32, g);
      acc = __builtin_amdgcn_wmma_f32_16x16x32_f16(false, a, false, b, (short)0, acc, false, false);
    }
    const int moff = g * 8;
#pragma unroll
    for (int r = 0; r < 8; ++r)
      sg[(r + moff) * G4 + nt * 16 + m] = acc[r];
  }
  __syncthreads();
  for (int i = tid; i < 16 * HH; i += 128) {
    int r = i >> 7, j = i & 127;
    int b = b0 + r;
    float gi = sg[r * G4 +          j] + bih[j]          + bhh[j];
    float gf = sg[r * G4 + HH  +    j] + bih[HH + j]     + bhh[HH + j];
    float gg = sg[r * G4 + 2 * HH + j] + bih[2 * HH + j] + bhh[2 * HH + j];
    float go = sg[r * G4 + 3 * HH + j] + bih[3 * HH + j] + bhh[3 * HH + j];
    float cv = sigm(gf) * c[b * HH + j] + sigm(gi) * tanhf(gg);
    float hv = sigm(go) * tanhf(cv);
    c[b * HH + j] = cv;
    h_h[b * HH + j] = (_Float16)hv;
    if (enc_out) enc_out[b * enc_stride + j] = (_Float16)hv;
  }
}

// ---------------------------------------------------------------------------
// ref_g / ref_p projection: out[b][h][s] = W @ enc[b][s] + bias. Big WMMA GEMM
// over 25600 rows. One block = 16 (b,s)-rows, 4 waves x 2 N-tiles.
// ---------------------------------------------------------------------------
__global__ __launch_bounds__(128)
void ref_proj_kernel(const _Float16* __restrict__ enc,  // (B*S) x 128 f16
                     const _Float16* __restrict__ W,    // 128 x 128 f16
                     const float* __restrict__ bias,
                     float* __restrict__ out) {         // (B,H,S)
  const int tid = threadIdx.x, lane = tid & 31, wave = tid >> 5;
  const int r0 = blockIdx.x * 16;
  const int g = lane >> 4, m = lane & 15;
  for (int nt = wave * 2; nt < wave * 2 + 2; ++nt) {
    v8f acc = {};
#pragma unroll
    for (int kt = 0; kt < 4; ++kt) {
      v16h a = frag_row(enc + (size_t)(r0 + m) * HH + kt * 32, g);
      v16h b = frag_row(W + (nt * 16 + m) * HH + kt * 32, g);
      acc = __builtin_amdgcn_wmma_f32_16x16x32_f16(false, a, false, b, (short)0, acc, false, false);
    }
    const int moff = g * 8;
    const int hcol = nt * 16 + m;
#pragma unroll
    for (int rr = 0; rr < 8; ++rr) {
      int r = r0 + rr + moff;
      int bb = r / SS, s = r % SS;
      out[((size_t)bb * HH + hcol) * SS + s] = acc[rr] + bias[hcol];
    }
  }
}

// ---------------------------------------------------------------------------
// glimpse: ql = h@Wq^T + b (WMMA); lg[b][s] = gV . tanh(ql + ref_g[:,s]);
// mask; softmax; q = ref_g @ p. One block = 16 batch rows.
// ---------------------------------------------------------------------------
__global__ __launch_bounds__(128)
void glimpse_kernel(const _Float16* __restrict__ h_h,
                    const _Float16* __restrict__ Wq, const float* __restrict__ bq,
                    const float* __restrict__ ref,   // (B,H,S)
                    const float* __restrict__ vvec,
                    unsigned char* __restrict__ mask,
                    const int* __restrict__ idx, int valid,
                    _Float16* __restrict__ q_out) {
  __shared__ float sql[16 * HH];
  __shared__ float slg[16 * 64];
  const int tid = threadIdx.x, lane = tid & 31, wave = tid >> 5;
  const int b0 = blockIdx.x * 16;
  const int g = lane >> 4, m = lane & 15;
  for (int nt = wave * 2; nt < wave * 2 + 2; ++nt) {
    v8f acc = {};
#pragma unroll
    for (int kt = 0; kt < 4; ++kt) {
      v16h a = frag_row(h_h + (size_t)(b0 + m) * HH + kt * 32, g);
      v16h b = frag_row(Wq + (nt * 16 + m) * HH + kt * 32, g);
      acc = __builtin_amdgcn_wmma_f32_16x16x32_f16(false, a, false, b, (short)0, acc, false, false);
    }
    const int moff = g * 8;
    const int hcol = nt * 16 + m;
#pragma unroll
    for (int rr = 0; rr < 8; ++rr)
      sql[(rr + moff) * HH + hcol] = acc[rr] + bq[hcol];
  }
  if (tid < 16 && valid) mask[(b0 + tid) * SS + idx[b0 + tid]] = 1;
  __syncthreads();
  for (int i = tid; i < 16 * SS; i += 128) {
    int r = i / SS, s = i % SS;
    int b = b0 + r;
    const float* rg = ref + (size_t)b * HH * SS + s;
    float acc = 0.f;
    for (int hh = 0; hh < HH; ++hh)
      acc += vvec[hh] * tanhf(sql[r * HH + hh] + rg[(size_t)hh * SS]);
    if (valid && mask[b * SS + s]) acc = NEGV;
    slg[r * 64 + s] = acc;
  }
  __syncthreads();
  if (tid < 16) {
    int r = tid;
    float mx = -INFINITY;
    for (int s = 0; s < SS; ++s) mx = fmaxf(mx, slg[r * 64 + s]);
    float sum = 0.f;
    for (int s = 0; s < SS; ++s) { float e = __expf(slg[r * 64 + s] - mx); slg[r * 64 + s] = e; sum += e; }
    float inv = 1.0f / sum;
    for (int s = 0; s < SS; ++s) slg[r * 64 + s] *= inv;
  }
  __syncthreads();
  for (int i = tid; i < 16 * HH; i += 128) {
    int r = i >> 7, hh = i & 127;
    int b = b0 + r;
    const float* rg = ref + (size_t)b * HH * SS + (size_t)hh * SS;
    float acc = 0.f;
    for (int s = 0; s < SS; ++s) acc += rg[s] * slg[r * 64 + s];
    q_out[b * HH + hh] = (_Float16)acc;
  }
}

// ---------------------------------------------------------------------------
// pointer: ql = q@Wq^T + b (WMMA); lg = C*tanh(pV . tanh(ql + ref_p)); mask;
// gumbel-argmax sample; softmax -> probs[idx]; dec_in = emb[b][idx].
// ---------------------------------------------------------------------------
__global__ __launch_bounds__(128)
void pointer_kernel(const _Float16* __restrict__ q_h,
                    const _Float16* __restrict__ Wq, const float* __restrict__ bq,
                    const float* __restrict__ ref, const float* __restrict__ vvec,
                    const unsigned char* __restrict__ mask, int valid, int t,
                    const _Float16* __restrict__ emb, _Float16* __restrict__ x_h,
                    int* __restrict__ idx, int* __restrict__ idx_seq,
                    float* __restrict__ probs_out, float* __restrict__ idxf_out) {
  __shared__ float sql[16 * HH];
  __shared__ float slg[16 * 64];
  __shared__ int   sidx[16];
  const int tid = threadIdx.x, lane = tid & 31, wave = tid >> 5;
  const int b0 = blockIdx.x * 16;
  const int g = lane >> 4, m = lane & 15;
  for (int nt = wave * 2; nt < wave * 2 + 2; ++nt) {
    v8f acc = {};
#pragma unroll
    for (int kt = 0; kt < 4; ++kt) {
      v16h a = frag_row(q_h + (size_t)(b0 + m) * HH + kt * 32, g);
      v16h b = frag_row(Wq + (nt * 16 + m) * HH + kt * 32, g);
      acc = __builtin_amdgcn_wmma_f32_16x16x32_f16(false, a, false, b, (short)0, acc, false, false);
    }
    const int moff = g * 8;
    const int hcol = nt * 16 + m;
#pragma unroll
    for (int rr = 0; rr < 8; ++rr)
      sql[(rr + moff) * HH + hcol] = acc[rr] + bq[hcol];
  }
  __syncthreads();
  for (int i = tid; i < 16 * SS; i += 128) {
    int r = i / SS, s = i % SS;
    int b = b0 + r;
    const float* rp = ref + (size_t)b * HH * SS + s;
    float acc = 0.f;
    for (int hh = 0; hh < HH; ++hh)
      acc += vvec[hh] * tanhf(sql[r * HH + hh] + rp[(size_t)hh * SS]);
    acc = CEXP * tanhf(acc);
    if (valid && mask[b * SS + s]) acc = NEGV;
    slg[r * 64 + s] = acc;
  }
  __syncthreads();
  if (tid < 16) {
    int r = tid, b = b0 + r;
    float mx = -INFINITY;
    for (int s = 0; s < SS; ++s) mx = fmaxf(mx, slg[r * 64 + s]);
    // gumbel-argmax sampling on logits (shift invariant)
    int best = 0; float bv = -INFINITY;
    for (int s = 0; s < SS; ++s) {
      uint32_t key = ((uint32_t)(t * SS + s) * 2654435761u) ^ ((uint32_t)b * 40503u + 0x9e3779b9u);
      float gum = -logf(-logf(u01(key)));
      float val = slg[r * 64 + s] + gum;
      if (val > bv) { bv = val; best = s; }
    }
    float sum = 0.f;
    for (int s = 0; s < SS; ++s) { float e = __expf(slg[r * 64 + s] - mx); slg[r * 64 + s] = e; sum += e; }
    float pb = slg[r * 64 + best] / sum;
    idx[b] = best;
    sidx[r] = best;
    idx_seq[t * BB + b] = best;
    probs_out[t * BB + b] = pb;
    idxf_out[t * BB + b] = (float)best;
  }
  __syncthreads();
  for (int i = tid; i < 16 * EE; i += 128) {
    int r = i >> 7, e = i & 127;
    int b = b0 + r;
    x_h[b * EE + e] = emb[((size_t)b * SS + sidx[r]) * EE + e];
  }
}

// ---------------------------------------------------------------------------
// final: actions from input gather + tour-length reward
// ---------------------------------------------------------------------------
__global__ void final_kernel(const float* __restrict__ input, const int* __restrict__ idx_seq,
                             float* __restrict__ reward_out, float* __restrict__ actions_out) {
  int b = blockIdx.x * blockDim.x + threadIdx.x;
  if (b >= BB) return;
  float fx = 0.f, fy = 0.f, px = 0.f, py = 0.f, rw = 0.f;
  for (int t = 0; t < SS; ++t) {
    int id = idx_seq[t * BB + b];
    float ax = input[b * 2 * SS + id];
    float ay = input[b * 2 * SS + SS + id];
    actions_out[(size_t)(t * BB + b) * 2 + 0] = ax;
    actions_out[(size_t)(t * BB + b) * 2 + 1] = ay;
    if (t == 0) { fx = ax; fy = ay; }
    else { float dx = ax - px, dy = ay - py; rw += sqrtf(dx * dx + dy * dy); }
    px = ax; py = ay;
  }
  float dx = px - fx, dy = py - fy;
  rw += sqrtf(dx * dx + dy * dy);
  reward_out[b] = rw;
}

// ---------------------------------------------------------------------------
extern "C" void kernel_launch(void* const* d_in, const int* in_sizes, int n_in,
                              void* d_out, int out_size, void* d_ws, size_t ws_size,
                              hipStream_t stream) {
  (void)in_sizes; (void)n_in; (void)out_size; (void)ws_size;
  const float* input  = (const float*)d_in[0];
  const float* Ew     = (const float*)d_in[1];
  const float* eWih   = (const float*)d_in[2];
  const float* eWhh   = (const float*)d_in[3];
  const float* ebih   = (const float*)d_in[4];
  const float* ebhh   = (const float*)d_in[5];
  const float* dWih   = (const float*)d_in[6];
  const float* dWhh   = (const float*)d_in[7];
  const float* dbih   = (const float*)d_in[8];
  const float* dbhh   = (const float*)d_in[9];
  const float* g0     = (const float*)d_in[10];
  const float* gWq_w  = (const float*)d_in[11];
  const float* gWq_b  = (const float*)d_in[12];
  const float* gWref_w= (const float*)d_in[13];
  const float* gWref_b= (const float*)d_in[14];
  const float* gV     = (const float*)d_in[15];
  const float* pWq_w  = (const float*)d_in[16];
  const float* pWq_b  = (const float*)d_in[17];
  const float* pWref_w= (const float*)d_in[18];
  const float* pWref_b= (const float*)d_in[19];
  const float* pV     = (const float*)d_in[20];

  char* ws = (char*)d_ws;
  size_t off = 0;
  auto alloc = [&](size_t bytes) -> char* {
    char* p = ws + off;
    off = (off + bytes + 255) & ~(size_t)255;
    return p;
  };
  _Float16* eWih_h  = (_Float16*)alloc((size_t)G4 * EE * 2);
  _Float16* eWhh_h  = (_Float16*)alloc((size_t)G4 * HH * 2);
  _Float16* dWih_h  = (_Float16*)alloc((size_t)G4 * EE * 2);
  _Float16* dWhh_h  = (_Float16*)alloc((size_t)G4 * HH * 2);
  _Float16* gWq_h   = (_Float16*)alloc((size_t)HH * HH * 2);
  _Float16* pWq_h   = (_Float16*)alloc((size_t)HH * HH * 2);
  _Float16* gWref_h = (_Float16*)alloc((size_t)HH * HH * 2);
  _Float16* pWref_h = (_Float16*)alloc((size_t)HH * HH * 2);
  _Float16* emb_h   = (_Float16*)alloc((size_t)BB * SS * EE * 2);
  _Float16* enc_h   = (_Float16*)alloc((size_t)BB * SS * HH * 2);
  _Float16* h_h     = (_Float16*)alloc((size_t)BB * HH * 2);
  _Float16* q_h     = (_Float16*)alloc((size_t)BB * HH * 2);
  _Float16* x_h     = (_Float16*)alloc((size_t)BB * EE * 2);
  float*    c_f     = (float*)   alloc((size_t)BB * HH * 4);
  float*    ref_g   = (float*)   alloc((size_t)BB * HH * SS * 4);
  float*    ref_p   = (float*)   alloc((size_t)BB * HH * SS * 4);
  unsigned char* mask = (unsigned char*)alloc((size_t)BB * SS);
  int*      idx     = (int*)     alloc((size_t)BB * 4);
  int*      idx_seq = (int*)     alloc((size_t)SS * BB * 4);

  float* fout = (float*)d_out;
  float* reward_out = fout;                       // [B]
  float* probs_out  = fout + BB;                  // [S,B]
  float* actions_out= fout + BB + SS * BB;        // [S,B,2]
  float* idxf_out   = fout + BB + SS * BB + SS * BB * 2; // [S,B]

  // weights -> f16
  f2h_kernel<<<(G4 * EE + 255) / 256, 256, 0, stream>>>(eWih, eWih_h, G4 * EE);
  f2h_kernel<<<(G4 * HH + 255) / 256, 256, 0, stream>>>(eWhh, eWhh_h, G4 * HH);
  f2h_kernel<<<(G4 * EE + 255) / 256, 256, 0, stream>>>(dWih, dWih_h, G4 * EE);
  f2h_kernel<<<(G4 * HH + 255) / 256, 256, 0, stream>>>(dWhh, dWhh_h, G4 * HH);
  f2h_kernel<<<(HH * HH + 255) / 256, 256, 0, stream>>>(gWq_w, gWq_h, HH * HH);
  f2h_kernel<<<(HH * HH + 255) / 256, 256, 0, stream>>>(pWq_w, pWq_h, HH * HH);
  f2h_kernel<<<(HH * HH + 255) / 256, 256, 0, stream>>>(gWref_w, gWref_h, HH * HH);
  f2h_kernel<<<(HH * HH + 255) / 256, 256, 0, stream>>>(pWref_w, pWref_h, HH * HH);

  emb_kernel<<<(BB * SS * EE + 255) / 256, 256, 0, stream>>>(input, Ew, emb_h);
  init_kernel<<<(BB * HH + 255) / 256, 256, 0, stream>>>(g0, h_h, c_f, x_h, mask, idx);

  // encoder: 50 sequential fused WMMA LSTM steps
  for (int t = 0; t < SS; ++t) {
    lstm_step_kernel<<<BB / 16, 128, 0, stream>>>(
        emb_h + (size_t)t * EE, SS * EE, eWih_h, eWhh_h, ebih, ebhh,
        c_f, h_h, enc_h + (size_t)t * HH, SS * HH);
  }

  // reference projections (big WMMA GEMMs)
  ref_proj_kernel<<<(BB * SS) / 16, 128, 0, stream>>>(enc_h, gWref_h, gWref_b, ref_g);
  ref_proj_kernel<<<(BB * SS) / 16, 128, 0, stream>>>(enc_h, pWref_h, pWref_b, ref_p);

  // decoder: 50 sequential steps, 3 fused kernels each
  for (int t = 0; t < SS; ++t) {
    lstm_step_kernel<<<BB / 16, 128, 0, stream>>>(
        x_h, EE, dWih_h, dWhh_h, dbih, dbhh, c_f, h_h, ((_Float16*)0), 0);
    glimpse_kernel<<<BB / 16, 128, 0, stream>>>(
        h_h, gWq_h, gWq_b, ref_g, gV, mask, idx, (t > 0) ? 1 : 0, q_h);
    pointer_kernel<<<BB / 16, 128, 0, stream>>>(
        q_h, pWq_h, pWq_b, ref_p, pV, mask, (t > 0) ? 1 : 0, t,
        emb_h, x_h, idx, idx_seq, probs_out, idxf_out);
  }

  final_kernel<<<2, 256, 0, stream>>>(input, idx_seq, reward_out, actions_out);
}
